// ConvNeXtBlock_77764677861410
// MI455X (gfx1250) — 8-bit matrix operand order repaired and checked by reading, compiled, not run on hardware
//
#include <hip/hip_runtime.h>
#include <hip/hip_fp16.h>

#define Bsz   8
#define DIM   512
#define Tlen  4096
#define INTER 1536
#define BT    (Bsz*Tlen)

typedef signed char i8;
typedef int v8i __attribute__((ext_vector_type(8)));
union V8 { v8i v; int i[8]; };

#if defined(__has_builtin)
#  if __has_builtin(__builtin_amdgcn_global_load_async_to_lds_b128)
#    define USE_ASYNC_LDS 1
#  endif
#endif
#if defined(USE_ASYNC_LDS)
#  if __has_builtin(__builtin_amdgcn_s_wait_asynccnt)
#    define WAIT_ASYNC() __builtin_amdgcn_s_wait_asynccnt(0)
#  else
#    define WAIT_ASYNC() asm volatile("s_wait_asynccnt 0x0" ::: "memory")
#  endif
typedef int v4i_vs __attribute__((__vector_size__(16)));
#  define ASYNC_LDS_B128(gsrc, ldst) \
     __builtin_amdgcn_global_load_async_to_lds_b128((v4i_vs*)(gsrc), (v4i_vs*)(ldst), 0, 0)
#endif

#define OFF_WSUMS 0
#define OFF_PART  256
#define OFF_AX    2048
#define OFF_DWQ   (OFF_AX + (size_t)BT*4)
#define OFF_W1Q   (OFF_DWQ + (size_t)DIM*7*4)
#define OFF_W2Q   (OFF_W1Q + (size_t)INTER*DIM)
#define OFF_YQ    (OFF_W2Q + (size_t)DIM*INTER)
#define OFF_YSC   (OFF_YQ  + (size_t)BT*DIM)
#define OFF_HQ    (OFF_YSC + (size_t)BT*4)
#define OFF_HSC   (OFF_HQ  + (size_t)BT*INTER)

#define WCHUNK 8192

__global__ __launch_bounds__(256) void wsum_stage1(const float* dw, const float* w1,
                                                   const float* w2, float* part) {
  const int bid = blockIdx.x;
  const float* p; int base, n;
  if (bid == 0)       { p = dw; base = 0;                 n = DIM*7;  }
  else if (bid <= 96) { p = w1; base = (bid-1)*WCHUNK;    n = WCHUNK; }
  else                { p = w2; base = (bid-97)*WCHUNK;   n = WCHUNK; }
  int tid = threadIdx.x;
  float s = 0.f;
  for (int i = tid; i < n; i += 256) s += fabsf(p[base + i]);
  __shared__ float r[256];
  r[tid] = s; __syncthreads();
  for (int o = 128; o > 0; o >>= 1) { if (tid < o) r[tid] += r[tid + o]; __syncthreads(); }
  if (tid == 0) part[bid] = r[0];
}

__global__ __launch_bounds__(64) void wsum_stage2(const float* part, float* wsums) {
  int t = threadIdx.x;
  if (t == 0) wsums[0] = part[0];
  else if (t == 1) { float s = 0.f; for (int i = 1;  i <= 96;  ++i) s += part[i]; wsums[1] = s; }
  else if (t == 2) { float s = 0.f; for (int i = 97; i <= 192; ++i) s += part[i]; wsums[2] = s; }
}

__global__ __launch_bounds__(256) void quantw_kernel(const float* dw, const float* w1,
                                                     const float* w2, const float* wsums,
                                                     float* dwq, i8* w1q, i8* w2q) {
  int idx = blockIdx.x * 256 + threadIdx.x;
  const int NDW = DIM*7, NW1 = INTER*DIM, NW2 = DIM*INTER;
  if (idx < NDW) {
    float s = fmaxf(wsums[0] / (float)NDW, 1e-5f);
    float q = fminf(fmaxf(rintf(dw[idx] / s), -1.f), 1.f);
    dwq[idx] = q * s;
  } else if (idx < NDW + NW1) {
    int i = idx - NDW;
    float s = fmaxf(wsums[1] / (float)NW1, 1e-5f);
    w1q[i] = (i8)fminf(fmaxf(rintf(w1[i] / s), -1.f), 1.f);
  } else if (idx < NDW + NW1 + NW2) {
    int i = idx - NDW - NW1;
    float s = fmaxf(wsums[2] / (float)NW2, 1e-5f);
    w2q[i] = (i8)fminf(fmaxf(rintf(w2[i] / s), -1.f), 1.f);
  }
}

__global__ __launch_bounds__(256) void absx_kernel(const float* x, float* ax) {
  int bb = blockIdx.y;
  int t  = blockIdx.x * 64 + (threadIdx.x & 63);
  int cg = threadIdx.x >> 6;
  __shared__ float red[4][64];
  float a = 0.f;
  for (int ci = 0; ci < 128; ++ci) {
    int c = cg * 128 + ci;
    a = fmaxf(a, fabsf(x[((size_t)bb*DIM + c)*Tlen + t]));
  }
  red[cg][threadIdx.x & 63] = a; __syncthreads();
  if (threadIdx.x < 64) {
    float m = fmaxf(fmaxf(red[0][threadIdx.x], red[1][threadIdx.x]),
                    fmaxf(red[2][threadIdx.x], red[3][threadIdx.x]));
    ax[bb*Tlen + blockIdx.x*64 + threadIdx.x] = m;
  }
}

__global__ __launch_bounds__(256) void conv_ln_quant_kernel(
    const float* x, const float* ax, const float* dwq, const float* dwb,
    const float* lng, const float* lnb, i8* yq, float* ysc_inv) {
  const int bb = blockIdx.y;
  const int t0 = blockIdx.x * 32;
  const int tid = threadIdx.x;
  const int tt = tid & 31;
  const int cg = tid >> 5;
  const int t  = t0 + tt;

  __shared__ __half ylds[DIM * 32];
  __shared__ float red[8][32], red2[8][32];
  __shared__ float mu_s[32], rs_s[32], sc_s[32];

  float svec[7], sinv[7]; bool inb[7];
  #pragma unroll
  for (int k = 0; k < 7; ++k) {
    int tg = t + k - 3;
    inb[k] = (tg >= 0 && tg < Tlen);
    float a = inb[k] ? ax[bb*Tlen + tg] : 1.f;
    float s = 127.f / fmaxf(a, 1e-5f);
    svec[k] = s; sinv[k] = 1.f / s;
  }

  float sum = 0.f, sumsq = 0.f;
  for (int ci = 0; ci < 64; ++ci) {
    int c = cg * 64 + ci;
    float acc = dwb[c];
    #pragma unroll
    for (int k = 0; k < 7; ++k) {
      if (inb[k]) {
        float xv = x[((size_t)bb*DIM + c)*Tlen + (t + k - 3)];
        float q  = fminf(fmaxf(rintf(xv * svec[k]), -128.f), 127.f);
        acc += (q * sinv[k]) * dwq[c*7 + k];
      }
    }
    ylds[c*32 + tt] = __float2half(acc);
    sum += acc; sumsq += acc * acc;
  }
  red[cg][tt] = sum; red2[cg][tt] = sumsq; __syncthreads();
  if (tid < 32) {
    float s = 0.f, s2 = 0.f;
    for (int w = 0; w < 8; ++w) { s += red[w][tid]; s2 += red2[w][tid]; }
    float mu = s * (1.f/DIM);
    float var = s2 * (1.f/DIM) - mu*mu;
    mu_s[tid] = mu; rs_s[tid] = rsqrtf(var + 1e-6f);
  }
  __syncthreads();
  float mu = mu_s[tt], rs = rs_s[tt];
  float am = 0.f;
  for (int ci = 0; ci < 64; ++ci) {
    int c = cg * 64 + ci;
    float yn = (__half2float(ylds[c*32 + tt]) - mu) * rs * lng[c] + lnb[c];
    am = fmaxf(am, fabsf(yn));
  }
  red[cg][tt] = am; __syncthreads();
  if (tid < 32) {
    float m = 0.f;
    for (int w = 0; w < 8; ++w) m = fmaxf(m, red[w][tid]);
    float sy = 127.f / fmaxf(m, 1e-5f);
    sc_s[tid] = sy;
    ysc_inv[bb*Tlen + t0 + tid] = 1.f / sy;
  }
  __syncthreads();
  float sy = sc_s[tt];
  for (int ci = 0; ci < 64; ++ci) {
    int c = cg * 64 + ci;
    float yn = (__half2float(ylds[c*32 + tt]) - mu) * rs * lng[c] + lnb[c];
    float q = fminf(fmaxf(rintf(yn * sy), -128.f), 127.f);
    yq[((size_t)bb*Tlen + t)*DIM + c] = (i8)q;
  }
}

__global__ __launch_bounds__(256) void gemm1_kernel(
    const i8* __restrict__ yq, const float* __restrict__ ysc_inv,
    const i8* __restrict__ w1q, const float* __restrict__ b1,
    const float* __restrict__ wsums, i8* __restrict__ hq, float* __restrict__ hsc_inv) {
  const int t0  = blockIdx.x * 16;
  const int tid = threadIdx.x;
  const int wv = tid >> 5, lane = tid & 31;
  const int m = lane & 15, hi = lane >> 4;

  __shared__ i8  ylds[16 * 516];
  __shared__ float stok[16], ssc[16], red[128];

#if defined(USE_ASYNC_LDS)
  for (int i = tid; i < 16*32; i += 256) {
    int row = i >> 5, seg = i & 31;
    ASYNC_LDS_B128(yq + (size_t)t0*DIM + row*DIM + seg*16,
                   &ylds[row*516 + seg*16]);
  }
  WAIT_ASYNC();
#else
  for (int i = tid; i < 16*128; i += 256) {
    int row = i >> 7, col = i & 127;
    *(int*)(&ylds[row*516 + col*4]) = ((const int*)(yq + (size_t)t0*DIM))[i];
  }
#endif
  if (tid < 16) stok[tid] = ysc_inv[t0 + tid];
  __syncthreads();

  const float sw1 = fmaxf(wsums[1] * (1.f/(float)(INTER*DIM)), 1e-5f);

  V8 acc[12];
  #pragma unroll
  for (int nt = 0; nt < 12; ++nt)
    #pragma unroll
    for (int v = 0; v < 8; ++v) acc[nt].i[v] = 0;

  for (int kk = 0; kk < 8; ++kk) {
    V8 a;
    #pragma unroll
    for (int v = 0; v < 8; ++v) {
      int Ks = ((v >> 1) << 4) + (hi << 3) + ((v & 1) << 2);
      a.i[v] = *(const int*)(&ylds[m*516 + kk*64 + Ks]);
    }
    #pragma unroll
    for (int nt = 0; nt < 12; ++nt) {
      int f = wv*192 + nt*16 + m;
      if (kk < 7) __builtin_prefetch(w1q + (size_t)f*DIM + (kk+1)*64, 0, 1);
      V8 bf;
      #pragma unroll
      for (int v = 0; v < 8; ++v) {
        int Ks = ((v >> 1) << 4) + (hi << 3) + ((v & 1) << 2);
        bf.i[v] = *(const int*)(w1q + (size_t)f*DIM + kk*64 + Ks);
      }
      acc[nt].v = __builtin_amdgcn_wmma_i32_16x16x64_iu8(
          true, a.v, true, bf.v, acc[nt].v, false, false);
    }
  }

  float hval[12][8];
  float tokmax[8];
  #pragma unroll
  for (int r = 0; r < 8; ++r) tokmax[r] = 0.f;
  #pragma unroll
  for (int nt = 0; nt < 12; ++nt) {
    int f = wv*192 + nt*16 + m;
    float bias = b1[f];
    #pragma unroll
    for (int r = 0; r < 8; ++r) {
      int tk = r + hi*8;
      float h = (float)acc[nt].i[r] * (sw1 * stok[tk]) + bias;
      h = 0.5f * h * (1.f + erff(h * 0.70710678118654752f));
      hval[nt][r] = h;
      tokmax[r] = fmaxf(tokmax[r], fabsf(h));
    }
  }
  #pragma unroll
  for (int off = 1; off < 16; off <<= 1)
    #pragma unroll
    for (int r = 0; r < 8; ++r)
      tokmax[r] = fmaxf(tokmax[r], __shfl_xor(tokmax[r], off, 32));
  if (m == 0) {
    #pragma unroll
    for (int r = 0; r < 8; ++r) red[wv*16 + r + hi*8] = tokmax[r];
  }
  __syncthreads();
  if (tid < 16) {
    float g = 0.f;
    for (int w = 0; w < 8; ++w) g = fmaxf(g, red[w*16 + tid]);
    float s = 127.f / fmaxf(g, 1e-5f);
    ssc[tid] = s;
    hsc_inv[t0 + tid] = 1.f / s;
  }
  __syncthreads();
  #pragma unroll
  for (int nt = 0; nt < 12; ++nt) {
    #pragma unroll
    for (int r = 0; r < 8; ++r) {
      int tk = r + hi*8;
      float q = fminf(fmaxf(rintf(hval[nt][r] * ssc[tk]), -128.f), 127.f);
      hq[(size_t)(t0 + tk)*INTER + wv*192 + nt*16 + m] = (i8)q;
    }
  }
}

#define SMEM2 (DIM*21*4)
__global__ __launch_bounds__(256) void gemm2_kernel(
    const i8* __restrict__ hq, const float* __restrict__ hsc_inv,
    const i8* __restrict__ w2q, const float* __restrict__ b2,
    const float* __restrict__ gamma, const float* __restrict__ wsums,
    const float* __restrict__ x, float* __restrict__ out) {
  const int t0  = blockIdx.x * 16;
  const int bb  = t0 >> 12;
  const int tt0 = t0 & (Tlen - 1);
  const int tid = threadIdx.x;
  const int wv = tid >> 5, lane = tid & 31;
  const int m = lane & 15, hi = lane >> 4;

  __shared__ char smem[SMEM2];
  __shared__ float stok[16];
  i8* hlds = (i8*)smem;

#if defined(USE_ASYNC_LDS)
  for (int i = tid; i < 16*96; i += 256) {
    int row = i / 96, seg = i - row*96;
    ASYNC_LDS_B128(hq + (size_t)t0*INTER + row*INTER + seg*16,
                   &hlds[row*1552 + seg*16]);
  }
  WAIT_ASYNC();
#else
  for (int i = tid; i < 16*384; i += 256) {
    int row = i / 384, col = i - row*384;
    *(int*)(&hlds[row*1552 + col*4]) = ((const int*)(hq + (size_t)t0*INTER))[i];
  }
#endif
  if (tid < 16) stok[tid] = hsc_inv[t0 + tid];
  __syncthreads();

  const float sw2 = fmaxf(wsums[2] * (1.f/(float)(DIM*INTER)), 1e-5f);

  V8 acc[4];
  #pragma unroll
  for (int nt = 0; nt < 4; ++nt)
    #pragma unroll
    for (int v = 0; v < 8; ++v) acc[nt].i[v] = 0;

  for (int kk = 0; kk < 24; ++kk) {
    V8 a;
    #pragma unroll
    for (int v = 0; v < 8; ++v) {
      int Ks = ((v >> 1) << 4) + (hi << 3) + ((v & 1) << 2);
      a.i[v] = *(const int*)(&hlds[m*1552 + kk*64 + Ks]);
    }
    #pragma unroll
    for (int nt = 0; nt < 4; ++nt) {
      int c = wv*64 + nt*16 + m;
      if (kk < 23) __builtin_prefetch(w2q + (size_t)c*INTER + (kk+1)*64, 0, 1);
      V8 bf;
      #pragma unroll
      for (int v = 0; v < 8; ++v) {
        int Ks = ((v >> 1) << 4) + (hi << 3) + ((v & 1) << 2);
        bf.i[v] = *(const int*)(w2q + (size_t)c*INTER + kk*64 + Ks);
      }
      acc[nt].v = __builtin_amdgcn_wmma_i32_16x16x64_iu8(
          true, a.v, true, bf.v, acc[nt].v, false, false);
    }
  }

  __syncthreads();
  float* olds = (float*)smem;
  #pragma unroll
  for (int nt = 0; nt < 4; ++nt) {
    int c = wv*64 + nt*16 + m;
    float bias = b2[c], gm = gamma[c];
    #pragma unroll
    for (int r = 0; r < 8; ++r) {
      int tk = r + hi*8;
      float o = (float)acc[nt].i[r] * (sw2 * stok[tk]) + bias;
      olds[c*21 + tk] = gm * o;
    }
  }
  __syncthreads();
  for (int i = tid; i < DIM*16; i += 256) {
    int row = i >> 4, col = i & 15;
    size_t gi = ((size_t)bb*DIM + row)*Tlen + tt0 + col;
    __builtin_nontemporal_store(x[gi] + olds[row*21 + col], &out[gi]);
  }
}

extern "C" void kernel_launch(void* const* d_in, const int* in_sizes, int n_in,
                              void* d_out, int out_size, void* d_ws, size_t ws_size,
                              hipStream_t stream) {
  (void)in_sizes; (void)n_in; (void)out_size; (void)ws_size;
  const float* x    = (const float*)d_in[0];
  const float* dw_w = (const float*)d_in[1];
  const float* dw_b = (const float*)d_in[2];
  const float* ln_g = (const float*)d_in[3];
  const float* ln_b = (const float*)d_in[4];
  const float* w1   = (const float*)d_in[5];
  const float* b1   = (const float*)d_in[6];
  const float* w2   = (const float*)d_in[7];
  const float* b2   = (const float*)d_in[8];
  const float* gmm  = (const float*)d_in[9];
  float* out = (float*)d_out;

  char* ws = (char*)d_ws;
  float* wsums = (float*)(ws + OFF_WSUMS);
  float* part  = (float*)(ws + OFF_PART);
  float* ax    = (float*)(ws + OFF_AX);
  float* dwq   = (float*)(ws + OFF_DWQ);
  i8*    w1q   = (i8*)(ws + OFF_W1Q);
  i8*    w2q   = (i8*)(ws + OFF_W2Q);
  i8*    yq    = (i8*)(ws + OFF_YQ);
  float* ysc   = (float*)(ws + OFF_YSC);
  i8*    hq    = (i8*)(ws + OFF_HQ);
  float* hsc   = (float*)(ws + OFF_HSC);

  wsum_stage1<<<193, 256, 0, stream>>>(dw_w, w1, w2, part);
  wsum_stage2<<<1, 64, 0, stream>>>(part, wsums);
  {
    int ntot = DIM*7 + INTER*DIM + DIM*INTER;
    quantw_kernel<<<(ntot + 255)/256, 256, 0, stream>>>(dw_w, w1, w2, wsums, dwq, w1q, w2q);
  }
  absx_kernel<<<dim3(Tlen/64, Bsz), 256, 0, stream>>>(x, ax);
  conv_ln_quant_kernel<<<dim3(Tlen/32, Bsz), 256, 0, stream>>>(x, ax, dwq, dw_b, ln_g, ln_b, yq, ysc);
  gemm1_kernel<<<BT/16, 256, 0, stream>>>(yq, ysc, w1q, b1, wsums, hq, hsc);
  gemm2_kernel<<<BT/16, 256, 0, stream>>>(hq, hsc, w2q, b2, gmm, wsums, x, out);
}
